// T2SModel_23751169147061
// MI455X (gfx1250) — matrix operand order repaired and checked by reading, not compiled, not run on hardware
//
#include <hip/hip_runtime.h>

#define D_      512
#define H_      16
#define HD_     32
#define FF_     2048
#define L_      24
#define VOCAB_  1025
#define XL_     512
#define YL_     512
#define SEQ_    1024
#define SMAX_   1040
#define ND_     15
#define EPS_    1e-5f
#define REPPEN_ 1.35f
#define SCALE_  0.17677669529663687f  /* 1/sqrt(32) */

typedef __attribute__((ext_vector_type(16))) _Float16 v16h;
typedef __attribute__((ext_vector_type(8)))  _Float16 v8h;
typedef __attribute__((ext_vector_type(8)))  float    v8f;

// ---------------------------------------------------------------------------
// Simple batched NT GEMM via WMMA (one 16x16 tile / wave). Used only for the
// P·V GEMM whose N=32 doesn't fit the 64x64 blocked kernel.
// ---------------------------------------------------------------------------
__global__ void gemm_wmma_nt(const _Float16* __restrict__ A,
                             const _Float16* __restrict__ W,
                             const float* __restrict__ bias,
                             float* __restrict__ Cf, _Float16* __restrict__ Ch,
                             int M, int N, int K, int lda, int ldw, int ldc,
                             long long sA, long long sW, long long sC,
                             float alpha, int relu) {
  int lane = threadIdx.x & 31;
  int wave = threadIdx.x >> 5;
  int tn = blockIdx.x * 4 + wave;
  int tm = blockIdx.y;
  int b  = blockIdx.z;
  if (tn * 16 >= N) return;            // uniform per-wave exit
  int m  = lane & 15;
  int hi = lane >> 4;
  const _Float16* arow = A + (long long)b * sA + (long long)(tm * 16 + m) * lda;
  const _Float16* wrow = W + (long long)b * sW + (long long)(tn * 16 + m) * ldw;
  v8f acc = {};
  for (int k0 = 0; k0 < K; k0 += 32) {
    v16h bv = *(const v16h*)(wrow + k0 + hi * 16);
    v16h av = *(const v16h*)(arow + k0 + hi * 16);
    acc = __builtin_amdgcn_wmma_f32_16x16x32_f16(false, av, false, bv,
                                                 (short)0, acc, false, false);
  }
  int col = tn * 16 + m;
  float bvv = bias ? bias[col] : 0.0f;
#pragma unroll
  for (int r = 0; r < 8; ++r) {
    int row = tm * 16 + r + 8 * hi;
    float v = acc[r] * alpha + bvv;
    if (relu && v < 0.f) v = 0.f;
    long long off = (long long)b * sC + (long long)row * ldc + col;
    if (Cf) Cf[off] = v;
    if (Ch) Ch[off] = (_Float16)v;
  }
}

// ---------------------------------------------------------------------------
// Register-blocked batched NT GEMM: each wave owns a 64x64 output block
// (4x4 tiles of 16x16 -> 16 independent accumulator chains, no WMMA->WMMA
// RAW hazards, 4x operand reuse from registers). Requires M%64==0; N blocks
// guarded per-wave; K%32==0.
// ---------------------------------------------------------------------------
__global__ void gemm_wmma_nt_blk(const _Float16* __restrict__ A,
                                 const _Float16* __restrict__ W,
                                 const float* __restrict__ bias,
                                 float* __restrict__ Cf, _Float16* __restrict__ Ch,
                                 int M, int N, int K, int lda, int ldw, int ldc,
                                 long long sA, long long sW, long long sC,
                                 float alpha, int relu) {
  int lane = threadIdx.x & 31;
  int wave = threadIdx.x >> 5;
  int tn = blockIdx.x * 4 + wave;      // 64-wide N block per wave
  int tm = blockIdx.y;                 // 64-tall M block
  int b  = blockIdx.z;
  if (tn * 64 >= N) return;            // uniform per-wave exit
  int m  = lane & 15;
  int hi = lane >> 4;
  const _Float16* Ab = A + (long long)b * sA;
  const _Float16* Wb = W + (long long)b * sW;
  const _Float16* arow[4];
  const _Float16* wrow[4];
#pragma unroll
  for (int i = 0; i < 4; ++i) {
    arow[i] = Ab + (long long)(tm * 64 + i * 16 + m) * lda;
    wrow[i] = Wb + (long long)(tn * 64 + i * 16 + m) * ldw;
  }
  v8f acc[4][4];
#pragma unroll
  for (int i = 0; i < 4; ++i)
#pragma unroll
    for (int j = 0; j < 4; ++j) acc[i][j] = (v8f){};
  for (int k0 = 0; k0 < K; k0 += 32) {
    v16h af[4], bf[4];
#pragma unroll
    for (int i = 0; i < 4; ++i) {
      af[i] = *(const v16h*)(arow[i] + k0 + hi * 16);
      bf[i] = *(const v16h*)(wrow[i] + k0 + hi * 16);
    }
#pragma unroll
    for (int i = 0; i < 4; ++i)
#pragma unroll
      for (int j = 0; j < 4; ++j)
        acc[i][j] = __builtin_amdgcn_wmma_f32_16x16x32_f16(false, af[i], false, bf[j],
                                                           (short)0, acc[i][j],
                                                           false, false);
  }
#pragma unroll
  for (int j = 0; j < 4; ++j) {
    int col = tn * 64 + j * 16 + m;
    float bvv = bias ? bias[col] : 0.0f;
#pragma unroll
    for (int i = 0; i < 4; ++i) {
#pragma unroll
      for (int r = 0; r < 8; ++r) {
        int row = tm * 64 + i * 16 + r + 8 * hi;
        float v = acc[i][j][r] * alpha + bvv;
        if (relu && v < 0.f) v = 0.f;
        long long off = (long long)b * sC + (long long)row * ldc + col;
        if (Cf) Cf[off] = v;
        if (Ch) Ch[off] = (_Float16)v;
      }
    }
  }
}

// ---------------------------------------------------------------------------
__global__ void cvt_f32_f16(const float* __restrict__ a, _Float16* __restrict__ d, long long n) {
  long long i = (long long)blockIdx.x * blockDim.x + threadIdx.x;
  if (i < n) d[i] = (_Float16)a[i];
}

__global__ void compute_pe(float* pe, int n, int d) {
  long long i = (long long)blockIdx.x * blockDim.x + threadIdx.x;
  if (i >= (long long)n * d) return;
  int p = (int)(i / d), c = (int)(i % d);
  int pair = c >> 1;
  float div = __expf((2.0f * pair) * (-logf(10000.0f) / (float)d));
  float ang = (float)p * div;
  pe[i] = (c & 1) ? __cosf(ang) : __sinf(ang);
}

__global__ void build_berth(const float* __restrict__ rb, const float* __restrict__ tb,
                            _Float16* __restrict__ o) {
  int i = blockIdx.x * 256 + threadIdx.x;
  if (i >= XL_ * 1024) return;
  int t = i >> 10, c = i & 1023;
  float v = (t < 256) ? rb[t * 1024 + c] : tb[(t - 256) * 1024 + c];
  o[i] = (_Float16)v;
}

__global__ void embed_x(const int* __restrict__ ref_seq, const int* __restrict__ text_seq,
                        const float* __restrict__ text_emb, const float* __restrict__ bertC,
                        const float* __restrict__ pex, const float* __restrict__ ax,
                        float* __restrict__ x, _Float16* __restrict__ xh) {
  int i = blockIdx.x * 256 + threadIdx.x;
  if (i >= XL_ * D_) return;
  int t = i >> 9, c = i & 511;
  int id = (t < 256) ? ref_seq[t] : text_seq[t - 256];
  float v = text_emb[id * D_ + c] + bertC[i] + ax[0] * pex[i];
  x[i] = v; xh[i] = (_Float16)v;
}

__global__ void embed_y(const int* __restrict__ prompts, const float* __restrict__ audio_emb,
                        const float* __restrict__ pey, const float* __restrict__ ay,
                        float* __restrict__ x, _Float16* __restrict__ xh) {
  int i = blockIdx.x * 256 + threadIdx.x;
  if (i >= YL_ * D_) return;
  int t = i >> 9, c = i & 511;
  float v = audio_emb[prompts[t] * D_ + c] + ay[0] * pey[t * D_ + c];
  x[XL_ * D_ + i] = v; xh[XL_ * D_ + i] = (_Float16)v;
}

__global__ void split_qkv(const float* __restrict__ qkv, _Float16* __restrict__ qh,
                          _Float16* __restrict__ Kc, _Float16* __restrict__ Vc,
                          _Float16* __restrict__ Vt) {
  int i = blockIdx.x * 256 + threadIdx.x;
  if (i >= SEQ_ * D_) return;
  int s = i >> 9, j = i & 511, h = j >> 5, d = j & 31;
  qh[((long long)h * SEQ_ + s) * HD_ + d] = (_Float16)qkv[s * (3 * D_) + j];
  Kc[((long long)h * SMAX_ + s) * HD_ + d] = (_Float16)qkv[s * (3 * D_) + D_ + j];
  float v = qkv[s * (3 * D_) + 2 * D_ + j];
  Vc[((long long)h * SMAX_ + s) * HD_ + d] = (_Float16)v;
  Vt[((long long)h * HD_ + d) * SEQ_ + s] = (_Float16)v;
}

// masked softmax, in place on fp16 scores [H][SEQ][SEQ]
__global__ void softmax_mask(_Float16* __restrict__ sc) {
  int q = blockIdx.x, h = blockIdx.y, tid = threadIdx.x;
  _Float16* row = sc + ((long long)h * SEQ_ + q) * SEQ_;
  __shared__ float buf[SEQ_];
  __shared__ float red[256];
  float mx = -1e30f;
  for (int c = tid; c < SEQ_; c += 256) {
    bool mask = (c >= XL_) && (q < XL_ || c > q);
    float v = mask ? -1e30f : (float)row[c];
    buf[c] = v; mx = fmaxf(mx, v);
  }
  red[tid] = mx; __syncthreads();
  for (int s = 128; s > 0; s >>= 1) { if (tid < s) red[tid] = fmaxf(red[tid], red[tid + s]); __syncthreads(); }
  mx = red[0]; __syncthreads();
  float sum = 0.f;
  for (int c = tid; c < SEQ_; c += 256) {
    float e = (buf[c] <= -1e29f) ? 0.f : __expf(buf[c] - mx);
    buf[c] = e; sum += e;
  }
  red[tid] = sum; __syncthreads();
  for (int s = 128; s > 0; s >>= 1) { if (tid < s) red[tid] += red[tid + s]; __syncthreads(); }
  float z = red[0];
  for (int c = tid; c < SEQ_; c += 256) row[c] = (_Float16)(buf[c] / z);
}

__global__ void merge_heads(const float* __restrict__ a, _Float16* __restrict__ o) {
  int i = blockIdx.x * 256 + threadIdx.x;
  if (i >= SEQ_ * D_) return;
  int s = i >> 9, j = i & 511, h = j >> 5, d = j & 31;
  o[i] = (_Float16)a[((long long)h * SEQ_ + s) * HD_ + d];
}

// x[row] = LN(x[row] + delta[row]) * w + b ; optional fp16 re-emit
__global__ void residual_ln(float* __restrict__ x, const float* __restrict__ delta,
                            const float* __restrict__ w, const float* __restrict__ b,
                            _Float16* __restrict__ xh) {
  int row = blockIdx.x, tid = threadIdx.x;
  __shared__ float y[D_];
  __shared__ float red[256];
  float s = 0.f;
  for (int c = tid; c < D_; c += 256) { float v = x[row * D_ + c] + delta[row * D_ + c]; y[c] = v; s += v; }
  red[tid] = s; __syncthreads();
  for (int k = 128; k > 0; k >>= 1) { if (tid < k) red[tid] += red[tid + k]; __syncthreads(); }
  float mean = red[0] / (float)D_; __syncthreads();
  s = 0.f;
  for (int c = tid; c < D_; c += 256) { float d2 = y[c] - mean; s += d2 * d2; }
  red[tid] = s; __syncthreads();
  for (int k = 128; k > 0; k >>= 1) { if (tid < k) red[tid] += red[tid + k]; __syncthreads(); }
  float inv = rsqrtf(red[0] / (float)D_ + EPS_);
  for (int c = tid; c < D_; c += 256) {
    float v = (y[c] - mean) * inv * w[c] + b[c];
    x[row * D_ + c] = v;
    if (xh) xh[row * D_ + c] = (_Float16)v;
  }
}

// y[n] = act( W[n,:] . x + bias[n] ) ; one wave per output row, coalesced
// b128 weight loads -> pure HBM-bandwidth-bound decode path.
__global__ void gemv_f16w(const _Float16* __restrict__ W, const float* __restrict__ x,
                          const float* __restrict__ bias, float* __restrict__ y,
                          int N, int K, int relu) {
  int lane = threadIdx.x & 31;
  int row = blockIdx.x * (blockDim.x >> 5) + (threadIdx.x >> 5);
  if (row >= N) return;                 // uniform per-wave exit
  const _Float16* w = W + (long long)row * K;
  float acc = 0.f;
  for (int k = lane * 8; k < K; k += 32 * 8) {
    v8h wv = *(const v8h*)(w + k);
#pragma unroll
    for (int t = 0; t < 8; ++t) acc += (float)wv[t] * x[k + t];
  }
#pragma unroll
  for (int o = 16; o > 0; o >>= 1) acc += __shfl_down(acc, o, 32);
  if (lane == 0) {
    if (bias) acc += bias[row];
    if (relu && acc < 0.f) acc = 0.f;
    y[row] = acc;
  }
}

__global__ void embed_xs(const int* __restrict__ tok, const float* __restrict__ audio_emb,
                         const float* __restrict__ pey, const float* __restrict__ ay,
                         int pe_row, float* __restrict__ xs) {
  int c = blockIdx.x * 256 + threadIdx.x;
  if (c >= D_) return;
  int id = tok[0];
  xs[c] = audio_emb[id * D_ + c] + ay[0] * pey[pe_row * D_ + c];
}

__global__ void append_kv(const float* __restrict__ qkv, _Float16* __restrict__ Kc,
                          _Float16* __restrict__ Vc, int pos) {
  int j = blockIdx.x * 256 + threadIdx.x;
  if (j >= D_) return;
  int h = j >> 5, d = j & 31;
  Kc[((long long)h * SMAX_ + pos) * HD_ + d] = (_Float16)qkv[D_ + j];
  Vc[((long long)h * SMAX_ + pos) * HD_ + d] = (_Float16)qkv[2 * D_ + j];
}

__global__ void decode_attn(const float* __restrict__ qkv, const _Float16* __restrict__ Kc,
                            const _Float16* __restrict__ Vc, float* __restrict__ attn,
                            int scur) {
  int h = blockIdx.x, tid = threadIdx.x;
  __shared__ float sc[SMAX_ + 16];
  __shared__ float red[256];
  const float* q = qkv + h * HD_;
  float mx = -1e30f;
  for (int s = tid; s < scur; s += 256) {
    const _Float16* kr = Kc + ((long long)h * SMAX_ + s) * HD_;
    float d = 0.f;
    for (int j = 0; j < HD_; ++j) d += q[j] * (float)kr[j];
    d *= SCALE_;
    sc[s] = d; mx = fmaxf(mx, d);
  }
  red[tid] = mx; __syncthreads();
  for (int k = 128; k > 0; k >>= 1) { if (tid < k) red[tid] = fmaxf(red[tid], red[tid + k]); __syncthreads(); }
  mx = red[0]; __syncthreads();
  float sum = 0.f;
  for (int s = tid; s < scur; s += 256) { float e = __expf(sc[s] - mx); sc[s] = e; sum += e; }
  red[tid] = sum; __syncthreads();
  for (int k = 128; k > 0; k >>= 1) { if (tid < k) red[tid] += red[tid + k]; __syncthreads(); }
  float z = red[0]; __syncthreads();
  for (int j = tid; j < HD_; j += 256) {
    float acc = 0.f;
    for (int s = 0; s < scur; ++s) acc += sc[s] * (float)Vc[((long long)h * SMAX_ + s) * HD_ + j];
    attn[h * HD_ + j] = acc / z;
  }
}

// ----- sampling: rep-penalty, top-k pivot, softmax, argmax(probs/noise) -----
__device__ inline float gauss_noise(unsigned step, unsigned v) {
  unsigned long long z = (((unsigned long long)(step + 1)) << 32) ^
                         (unsigned long long)(v * 2654435761u + 1234567u);
  z = (z ^ (z >> 30)) * 0xBF58476D1CE4E5B9ull;
  z = (z ^ (z >> 27)) * 0x94D049BB133111EBull;
  z ^= z >> 31;
  unsigned u1 = (unsigned)z, u2 = (unsigned)(z >> 32);
  float f1 = ((float)u1 + 1.0f) * (1.0f / 4294967296.0f);
  float f2 = (float)u2 * (1.0f / 4294967296.0f);
  return sqrtf(-2.0f * logf(f1)) * __cosf(6.28318530718f * f2);
}

__global__ void sample_kernel(const float* __restrict__ logits, const int* __restrict__ prompts,
                              int nprompt, const int* __restrict__ toks, int ntok, int step,
                              const int* __restrict__ topk, int* __restrict__ tok_out,
                              float* __restrict__ tokf_out) {
  const int V = VOCAB_;
  __shared__ float sl[1056];
  __shared__ float st[1056];
  __shared__ unsigned char fl[1056];
  __shared__ float rv[256];
  __shared__ int   ri[256];
  int tid = threadIdx.x;
  for (int v = tid; v < V; v += 256) { sl[v] = logits[v]; fl[v] = 0; }
  __syncthreads();
  for (int p = tid; p < nprompt + ntok; p += 256) {
    int t = (p < nprompt) ? prompts[p] : toks[p - nprompt];
    fl[t] = 1;
  }
  __syncthreads();
  for (int v = tid; v < V; v += 256) {
    if (fl[v]) { float s = sl[v]; sl[v] = (s < 0.f) ? s * REPPEN_ : s / REPPEN_; }
    st[v] = sl[v];
  }
  __syncthreads();
  int tk = topk[0]; if (tk > V) tk = V; if (tk < 1) tk = 1;
  float vmax = 0.f, pivot = 0.f;
  for (int it = 0; it < tk; ++it) {
    float bv = -1e38f; int bi = 0;
    for (int v = tid; v < V; v += 256) if (st[v] > bv) { bv = st[v]; bi = v; }
    rv[tid] = bv; ri[tid] = bi; __syncthreads();
    for (int s = 128; s > 0; s >>= 1) {
      if (tid < s && (rv[tid + s] > rv[tid] ||
                      (rv[tid + s] == rv[tid] && ri[tid + s] < ri[tid]))) {
        rv[tid] = rv[tid + s]; ri[tid] = ri[tid + s];
      }
      __syncthreads();
    }
    if (it == 0) vmax = rv[0];
    pivot = rv[0];
    if (tid == 0) st[ri[0]] = -1e38f;
    __syncthreads();
  }
  float sum = 0.f;
  for (int v = tid; v < V; v += 256) {
    float p = (sl[v] >= pivot) ? __expf(sl[v] - vmax) : 0.f;
    st[v] = p; sum += p;
  }
  rv[tid] = sum; __syncthreads();
  for (int s = 128; s > 0; s >>= 1) { if (tid < s) rv[tid] += rv[tid + s]; __syncthreads(); }
  float z = rv[0]; __syncthreads();
  float bv = -1e38f; int bi = 0;
  for (int v = tid; v < V; v += 256) {
    float q = gauss_noise((unsigned)step, (unsigned)v);
    float val = (st[v] / z) / q;
    if (val > bv) { bv = val; bi = v; }
  }
  rv[tid] = bv; ri[tid] = bi; __syncthreads();
  for (int s = 128; s > 0; s >>= 1) {
    if (tid < s && (rv[tid + s] > rv[tid] ||
                    (rv[tid + s] == rv[tid] && ri[tid + s] < ri[tid]))) {
      rv[tid] = rv[tid + s]; ri[tid] = ri[tid + s];
    }
    __syncthreads();
  }
  if (tid == 0) { tok_out[0] = ri[0]; tokf_out[0] = (float)ri[0]; }
}

// ---------------------------------------------------------------------------
extern "C" void kernel_launch(void* const* d_in, const int* in_sizes, int n_in,
                              void* d_out, int out_size, void* d_ws, size_t ws_size,
                              hipStream_t stream) {
  (void)in_sizes; (void)n_in; (void)out_size;
  const int*   prompts     = (const int*)d_in[0];
  const int*   ref_seq     = (const int*)d_in[1];
  const int*   text_seq    = (const int*)d_in[2];
  const float* ref_bert    = (const float*)d_in[3];
  const float* text_bert   = (const float*)d_in[4];
  const int*   top_k       = (const int*)d_in[5];
  const float* text_emb    = (const float*)d_in[6];
  const float* audio_emb   = (const float*)d_in[7];
  const float* bert_proj_w = (const float*)d_in[8];
  const float* bert_proj_b = (const float*)d_in[9];
  const float* alpha_x     = (const float*)d_in[10];
  const float* alpha_y     = (const float*)d_in[11];
  const float* qkv_w       = (const float*)d_in[12];
  const float* qkv_b       = (const float*)d_in[13];
  const float* out_w       = (const float*)d_in[14];
  const float* out_b       = (const float*)d_in[15];
  const float* w1          = (const float*)d_in[16];
  const float* b1          = (const float*)d_in[17];
  const float* w2          = (const float*)d_in[18];
  const float* b2          = (const float*)d_in[19];
  const float* ln1_w       = (const float*)d_in[20];
  const float* ln1_b       = (const float*)d_in[21];
  const float* ln2_w       = (const float*)d_in[22];
  const float* ln2_b       = (const float*)d_in[23];
  const float* pred_w      = (const float*)d_in[24];
  const float* pred_b      = (const float*)d_in[25];

  // ---- workspace carve (256B aligned bump allocator) ----
  char* base = (char*)d_ws;
  size_t off = 0;
  auto carve = [&](size_t bytes) -> void* {
    off = (off + 255) & ~(size_t)255;
    void* p = base + off; off += bytes; return p;
  };
  _Float16* qkvWh  = (_Float16*)carve((size_t)L_ * 3 * D_ * D_ * 2);
  _Float16* outWh  = (_Float16*)carve((size_t)L_ * D_ * D_ * 2);
  _Float16* w1h    = (_Float16*)carve((size_t)L_ * FF_ * D_ * 2);
  _Float16* w2h    = (_Float16*)carve((size_t)L_ * D_ * FF_ * 2);
  _Float16* predWh = (_Float16*)carve((size_t)VOCAB_ * D_ * 2);
  _Float16* bprojh = (_Float16*)carve((size_t)D_ * 1024 * 2);
  _Float16* berth  = (_Float16*)carve((size_t)XL_ * 1024 * 2);
  float*    pex    = (float*)carve((size_t)XL_ * D_ * 4);
  float*    pey    = (float*)carve((size_t)(YL_ + ND_ + 1) * D_ * 4);
  float*    x      = (float*)carve((size_t)SEQ_ * D_ * 4);
  _Float16* xh     = (_Float16*)carve((size_t)SEQ_ * D_ * 2);
  float*    delta  = (float*)carve((size_t)SEQ_ * D_ * 4);
  float*    qkvb   = (float*)carve((size_t)SEQ_ * 3 * D_ * 4);
  _Float16* qh     = (_Float16*)carve((size_t)H_ * SEQ_ * HD_ * 2);
  _Float16* Kc     = (_Float16*)carve((size_t)L_ * H_ * SMAX_ * HD_ * 2);
  _Float16* Vc     = (_Float16*)carve((size_t)L_ * H_ * SMAX_ * HD_ * 2);
  _Float16* Vt     = (_Float16*)carve((size_t)H_ * HD_ * SEQ_ * 2);
  _Float16* scores = (_Float16*)carve((size_t)H_ * SEQ_ * SEQ_ * 2);
  float*    attnO  = (float*)carve((size_t)H_ * SEQ_ * HD_ * 4);
  _Float16* attnOh = (_Float16*)carve((size_t)SEQ_ * D_ * 2);
  _Float16* ffh    = (_Float16*)carve((size_t)SEQ_ * FF_ * 2);
  int*      tokens = (int*)carve(64 * 4);
  float*    xs     = (float*)carve(D_ * 4);
  float*    qkvs   = (float*)carve(3 * D_ * 4);
  float*    attns  = (float*)carve(D_ * 4);
  float*    deltas = (float*)carve(D_ * 4);
  float*    ffs    = (float*)carve(FF_ * 4);
  if (off > ws_size) return;

  // blocked GEMM: 64x64 per wave, 4 waves/block
  auto gemmB = [&](const _Float16* A, const _Float16* W, const float* bias,
                   float* Cf, _Float16* Ch, int M, int N, int K,
                   int lda, int ldw, int ldc, long long sA, long long sW, long long sC,
                   int batch, float alpha, int relu) {
    dim3 g((N / 64 + 3) / 4, M / 64, batch);
    gemm_wmma_nt_blk<<<g, dim3(128), 0, stream>>>(A, W, bias, Cf, Ch, M, N, K,
                                                  lda, ldw, ldc, sA, sW, sC, alpha, relu);
  };
  // simple GEMM (16x16 per wave) for small-N cases
  auto gemmS = [&](const _Float16* A, const _Float16* W, const float* bias,
                   float* Cf, _Float16* Ch, int M, int N, int K,
                   int lda, int ldw, int ldc, long long sA, long long sW, long long sC,
                   int batch, float alpha, int relu) {
    dim3 g((N / 16 + 3) / 4, M / 16, batch);
    gemm_wmma_nt<<<g, dim3(128), 0, stream>>>(A, W, bias, Cf, Ch, M, N, K,
                                              lda, ldw, ldc, sA, sW, sC, alpha, relu);
  };
  auto cvt = [&](const float* s, _Float16* d, long long n) {
    cvt_f32_f16<<<(unsigned)((n + 255) / 256), 256, 0, stream>>>(s, d, n);
  };
  auto gemv = [&](const _Float16* W, const float* xv, const float* b, float* y,
                  int N, int K, int relu) {
    gemv_f16w<<<(N + 7) / 8, 256, 0, stream>>>(W, xv, b, y, N, K, relu);
  };

  // ---- weight conversion to fp16 (every call; deterministic) ----
  cvt(qkv_w, qkvWh, (long long)L_ * 3 * D_ * D_);
  cvt(out_w, outWh, (long long)L_ * D_ * D_);
  cvt(w1, w1h, (long long)L_ * FF_ * D_);
  cvt(w2, w2h, (long long)L_ * D_ * FF_);
  cvt(pred_w, predWh, (long long)VOCAB_ * D_);
  cvt(bert_proj_w, bprojh, (long long)D_ * 1024);

  // ---- positional encodings + embeddings ----
  compute_pe<<<(XL_ * D_ + 255) / 256, 256, 0, stream>>>(pex, XL_, D_);
  compute_pe<<<((YL_ + ND_ + 1) * D_ + 255) / 256, 256, 0, stream>>>(pey, YL_ + ND_ + 1, D_);
  build_berth<<<(XL_ * 1024 + 255) / 256, 256, 0, stream>>>(ref_bert, text_bert, berth);
  // bert projection: [512 x 1024] x [512 x 1024]^T -> delta[0:512]
  gemmB(berth, bprojh, bert_proj_b, delta, nullptr, XL_, D_, 1024, 1024, 1024, D_,
        0, 0, 0, 1, 1.0f, 0);
  embed_x<<<(XL_ * D_ + 255) / 256, 256, 0, stream>>>(ref_seq, text_seq, text_emb,
                                                      delta, pex, alpha_x, x, xh);
  embed_y<<<(YL_ * D_ + 255) / 256, 256, 0, stream>>>(prompts, audio_emb, pey, alpha_y, x, xh);

  // ---- prefill: 24 layers ----
  for (int l = 0; l < L_; ++l) {
    const _Float16* qw = qkvWh + (size_t)l * 3 * D_ * D_;
    const _Float16* ow = outWh + (size_t)l * D_ * D_;
    const _Float16* f1 = w1h + (size_t)l * FF_ * D_;
    const _Float16* f2 = w2h + (size_t)l * D_ * FF_;
    _Float16* Kcl = Kc + (size_t)l * H_ * SMAX_ * HD_;
    _Float16* Vcl = Vc + (size_t)l * H_ * SMAX_ * HD_;

    gemmB(xh, qw, qkv_b + l * 3 * D_, qkvb, nullptr, SEQ_, 3 * D_, D_,
          D_, D_, 3 * D_, 0, 0, 0, 1, 1.0f, 0);
    split_qkv<<<(SEQ_ * D_ + 255) / 256, 256, 0, stream>>>(qkvb, qh, Kcl, Vcl, Vt);
    // scores = Q (S x 32) . K^T per head, scaled, fp16 out
    gemmB(qh, Kcl, nullptr, nullptr, scores, SEQ_, SEQ_, HD_, HD_, HD_, SEQ_,
          (long long)SEQ_ * HD_, (long long)SMAX_ * HD_, (long long)SEQ_ * SEQ_,
          H_, SCALE_, 0);
    softmax_mask<<<dim3(SEQ_, H_), 256, 0, stream>>>(scores);
    // O = P (S x S) . Vt^T (32 x S)^T per head  (N=32 -> simple kernel)
    gemmS(scores, Vt, nullptr, attnO, nullptr, SEQ_, HD_, SEQ_, SEQ_, SEQ_, HD_,
          (long long)SEQ_ * SEQ_, (long long)HD_ * SEQ_, (long long)SEQ_ * HD_,
          H_, 1.0f, 0);
    merge_heads<<<(SEQ_ * D_ + 255) / 256, 256, 0, stream>>>(attnO, attnOh);
    gemmB(attnOh, ow, out_b + l * D_, delta, nullptr, SEQ_, D_, D_, D_, D_, D_,
          0, 0, 0, 1, 1.0f, 0);
    residual_ln<<<SEQ_, 256, 0, stream>>>(x, delta, ln1_w + l * D_, ln1_b + l * D_, xh);
    gemmB(xh, f1, b1 + l * FF_, nullptr, ffh, SEQ_, FF_, D_, D_, D_, FF_,
          0, 0, 0, 1, 1.0f, 1);
    gemmB(ffh, f2, b2 + l * D_, delta, nullptr, SEQ_, D_, FF_, FF_, FF_, D_,
          0, 0, 0, 1, 1.0f, 0);
    residual_ln<<<SEQ_, 256, 0, stream>>>(x, delta, ln2_w + l * D_, ln2_b + l * D_, xh);
  }

  float* outf = (float*)d_out;
  float* tokf = outf + 16 * VOCAB_;

  // ---- first logits + sample ----
  gemv(predWh, x + (size_t)(SEQ_ - 1) * D_, pred_b, outf, VOCAB_, D_, 0);
  sample_kernel<<<1, 256, 0, stream>>>(outf, prompts, YL_, tokens, 0, 0, top_k,
                                       tokens, tokf);

  // ---- 15 decode steps ----
  for (int i = 1; i <= ND_; ++i) {
    int dstep = i - 1;
    int pos = SEQ_ + dstep;
    int scur = pos + 1;
    embed_xs<<<2, 256, 0, stream>>>(tokens + (i - 1), audio_emb, pey, alpha_y,
                                    YL_ + dstep, xs);
    for (int l = 0; l < L_; ++l) {
      const _Float16* qw = qkvWh + (size_t)l * 3 * D_ * D_;
      const _Float16* ow = outWh + (size_t)l * D_ * D_;
      const _Float16* f1 = w1h + (size_t)l * FF_ * D_;
      const _Float16* f2 = w2h + (size_t)l * D_ * FF_;
      _Float16* Kcl = Kc + (size_t)l * H_ * SMAX_ * HD_;
      _Float16* Vcl = Vc + (size_t)l * H_ * SMAX_ * HD_;

      gemv(qw, xs, qkv_b + l * 3 * D_, qkvs, 3 * D_, D_, 0);
      append_kv<<<2, 256, 0, stream>>>(qkvs, Kcl, Vcl, pos);
      decode_attn<<<H_, 256, 0, stream>>>(qkvs, Kcl, Vcl, attns, scur);
      gemv(ow, attns, out_b + l * D_, deltas, D_, D_, 0);
      residual_ln<<<1, 256, 0, stream>>>(xs, deltas, ln1_w + l * D_, ln1_b + l * D_, nullptr);
      gemv(f1, xs, b1 + l * FF_, ffs, FF_, D_, 1);
      gemv(f2, ffs, b2 + l * D_, deltas, D_, FF_, 0);
      residual_ln<<<1, 256, 0, stream>>>(xs, deltas, ln2_w + l * D_, ln2_b + l * D_, nullptr);
    }
    gemv(predWh, xs, pred_b, outf + (size_t)i * VOCAB_, VOCAB_, D_, 0);
    sample_kernel<<<1, 256, 0, stream>>>(outf + (size_t)i * VOCAB_, prompts, YL_,
                                         tokens, i, i, top_k, tokens + i, tokf + i);
  }
}
